// AttentionModule_9088150798574
// MI455X (gfx1250) — compile-verified, matrix operand order checked by reading, not run on hardware
//
#include <hip/hip_runtime.h>
#include <hip/hip_bf16.h>
#include <math.h>

// CDNA5 wave32 WMMA types
typedef __attribute__((ext_vector_type(16))) _Float16 v16h;
typedef __attribute__((ext_vector_type(8)))  float    v8f;

#define HEADS 8
#define INV_SQRT_D 0.35355339059327373f  // 1/sqrt(8)

// Order-preserving float->uint encoding so we can use native u32 atomicMax
// (GLOBAL_ATOMIC_MAX_U32) instead of a CAS loop for the segment max.
__device__ __forceinline__ unsigned ordEncode(float f) {
  unsigned u = __float_as_uint(f);
  return (u & 0x80000000u) ? ~u : (u | 0x80000000u);
}
__device__ __forceinline__ float ordDecode(unsigned u) {
  return __uint_as_float((u & 0x80000000u) ? (u & 0x7fffffffu) : ~u);
}
#define ENC_NEG_INF 0x007FFFFFu  // ordEncode(-inf)

// ---------------------------------------------------------------------------
// Pass 0: init output (zeros) and per-node stats (-inf max, 0 sum)
// ---------------------------------------------------------------------------
__global__ void gat_init(float* __restrict__ out, int out_n,
                         unsigned* __restrict__ segmax,
                         float* __restrict__ segsum, int nh) {
  int i = blockIdx.x * blockDim.x + threadIdx.x;
  if (i < out_n) out[i] = 0.0f;
  if (i < nh) { segmax[i] = ENC_NEG_INF; segsum[i] = 0.0f; }
}

// ---------------------------------------------------------------------------
// Pass 1: logits via V_WMMA_F32_16X16X32_F16.
// Each wave handles 16 (edge,head) pairs = 2 edges x 8 heads in one WMMA.
// Lane i (i<16): pair i -> A row i = key(e,h) in K=0..7 (halves 0..7),
//                         B col i = query(dst[e],h) in K=0..7. Rest zero.
// D[i][i] = key . query  (K padded 8->32 with zeros).
// ---------------------------------------------------------------------------
__global__ void gat_logits_wmma(const float* __restrict__ qv,
                                const float* __restrict__ qs,
                                const float* __restrict__ kv,
                                const float* __restrict__ ks,
                                const int*   __restrict__ dst,
                                float*       __restrict__ logits,
                                unsigned*    __restrict__ segmax,
                                int E) {
  __shared__ float diag[8][256];  // 8 waves x 32 lanes x 8 accum VGPRs

  const int lane = threadIdx.x & 31;
  const int wave = threadIdx.x >> 5;
  const int e    = blockIdx.x * 16 + wave * 2 + (lane >> 3);
  const int h    = lane & 7;
  const bool active = (lane < 16) && (e < E);

  v16h a = {};   // A-matrix: 16x32 f16, lane<16 holds row M=lane, halves 0..7=K0..7
  v16h b = {};   // B-matrix: 32x16 f16, lane<16 holds col N=lane, halves 0..7=K0..7
  int n = 0;
  if (active) {
    n = dst[e];
    const float* kvp = kv + (size_t)e * 48 + 6 * h;
    const float* ksp = ks + (size_t)e * 16 + 2 * h;
    const float* qvp = qv + (size_t)n * 48 + 6 * h;
    const float* qsp = qs + (size_t)n * 16 + 2 * h;
#pragma unroll
    for (int j = 0; j < 6; ++j) {
      a[j] = (_Float16)kvp[j];
      b[j] = (_Float16)qvp[j];
    }
    a[6] = (_Float16)ksp[0]; a[7] = (_Float16)ksp[1];
    b[6] = (_Float16)qsp[0]; b[7] = (_Float16)qsp[1];
  }

  // EXEC must be all ones here: every lane executes the WMMA (inactive lanes
  // contribute zeros).  D = A x B + 0.
  v8f c = {};
  c = __builtin_amdgcn_wmma_f32_16x16x32_f16(
      /*neg_a=*/false, a, /*neg_b=*/false, b,
      /*c_mod=*/(short)0, c, /*reuse_a=*/false, /*reuse_b=*/false);

  // Extract diagonal through LDS.  C/D layout: VGPR r, lanes 0-15 -> (M=r,
  // N=lane); lanes 16-31 -> (M=r+8, N=lane-16).
  float* wl = &diag[wave][0];
#pragma unroll
  for (int r = 0; r < 8; ++r) wl[lane * 8 + r] = c[r];
  __syncthreads();

  if (active) {
    const int srcLane = (lane < 8) ? lane : (lane + 16);
    const int srcV    = (lane < 8) ? lane : (lane - 8);
    float logit = wl[srcLane * 8 + srcV] * INV_SQRT_D;
    logits[(size_t)e * HEADS + h] = logit;
    atomicMax(&segmax[(size_t)n * HEADS + h], ordEncode(logit));
  }
}

// ---------------------------------------------------------------------------
// Pass 2: ex = exp(logit - seg_max[dst]);  seg_sum += ex  (in-place on logits)
// ---------------------------------------------------------------------------
__global__ void gat_softmax_norm(const int* __restrict__ dst,
                                 float* __restrict__ logits_ex,
                                 const unsigned* __restrict__ segmax,
                                 float* __restrict__ segsum, int EH) {
  int t = blockIdx.x * blockDim.x + threadIdx.x;
  if (t >= EH) return;
  int e = t >> 3, h = t & 7;
  int n = dst[e];
  float m  = ordDecode(segmax[(size_t)n * HEADS + h]);
  float ex = __expf(logits_ex[t] - m);
  logits_ex[t] = ex;
  atomicAdd(&segsum[(size_t)n * HEADS + h], ex);
}

// ---------------------------------------------------------------------------
// Pass 3: attn = ex / seg_sum[dst]; scatter-accumulate attn * v into nodes.
// Outputs (25.6 MB) stay L2-resident -> atomics run at L2 rate.
// ---------------------------------------------------------------------------
__global__ void gat_aggregate(const int* __restrict__ dst,
                              const float* __restrict__ ex,
                              const float* __restrict__ segsum,
                              const float* __restrict__ vv,
                              const float* __restrict__ vs,
                              float* __restrict__ outv,
                              float* __restrict__ outs, int EH) {
  int t = blockIdx.x * blockDim.x + threadIdx.x;
  if (t >= EH) return;
  int e = t >> 3, h = t & 7;
  int n = dst[e];
  float attn = ex[t] / segsum[(size_t)n * HEADS + h];

  const float* vvp = vv + (size_t)e * 48 + 6 * h;
  float*       ovp = outv + (size_t)n * 48 + 6 * h;
#pragma unroll
  for (int j = 0; j < 6; ++j) atomicAdd(&ovp[j], attn * vvp[j]);

  const float* vsp = vs + (size_t)e * 16 + 2 * h;
  float*       osp = outs + (size_t)n * 16 + 2 * h;
  atomicAdd(&osp[0], attn * vsp[0]);
  atomicAdd(&osp[1], attn * vsp[1]);
}

// ---------------------------------------------------------------------------
extern "C" void kernel_launch(void* const* d_in, const int* in_sizes, int n_in,
                              void* d_out, int out_size, void* d_ws, size_t ws_size,
                              hipStream_t stream) {
  const float* qv  = (const float*)d_in[0];  // [N,16,3]
  const float* qs  = (const float*)d_in[1];  // [N,16,1]
  const float* kv  = (const float*)d_in[2];  // [E,16,3]
  const float* ks  = (const float*)d_in[3];  // [E,16,1]
  const float* vv  = (const float*)d_in[4];  // [E,16,3]
  const float* vs  = (const float*)d_in[5];  // [E,16,1]
  const int*   dst = (const int*)  d_in[6];  // [E]

  const int N  = in_sizes[0] / 48;
  const int E  = in_sizes[2] / 48;
  const int NH = N * HEADS;
  const int EH = E * HEADS;

  // Workspace: logits/ex [E*H] f32 | segmax [N*H] u32 | segsum [N*H] f32
  float*    logits = (float*)d_ws;
  unsigned* segmax = (unsigned*)(logits + (size_t)EH);
  float*    segsum = (float*)(segmax + (size_t)NH);

  float* out  = (float*)d_out;
  float* outv = out;                    // [N,16,3] flat = N*48
  float* outs = out + (size_t)N * 48;   // [N,16,1] flat = N*16

  gat_init<<<(out_size + 255) / 256, 256, 0, stream>>>(out, out_size, segmax, segsum, NH);

  // 16 edges per block (8 waves x 2 edges), one WMMA per wave.
  gat_logits_wmma<<<(E + 15) / 16, 256, 0, stream>>>(qv, qs, kv, ks, dst,
                                                     logits, segmax, E);

  gat_softmax_norm<<<(EH + 255) / 256, 256, 0, stream>>>(dst, logits, segmax,
                                                         segsum, EH);

  gat_aggregate<<<(EH + 255) / 256, 256, 0, stream>>>(dst, logits, segsum,
                                                      vv, vs, outv, outs, EH);
}
